// RecurrentStateSpaceModel_85255100826103
// MI455X (gfx1250) — matrix operand order repaired and checked by reading, not compiled, not run on hardware
//
#include <hip/hip_runtime.h>
#include <math.h>

// ---------------- problem constants ----------------
#define BB 512
#define TT 100
#define STATE 200
#define LATENT 30
#define EMBED 1024
#define ACT 6

// padded dims
#define SKP  224   // state K padded (7*32)
#define XAK  64    // concat(s,a)=36 -> K padded (2*32)
#define QKP  1248  // concat(h,e)=1224 -> K padded (39*32)
#define NT_S 13
#define KT_S 7
#define KT_XA 2
#define KT_Q 39

typedef __bf16 bf16_t;
typedef __attribute__((ext_vector_type(16))) __bf16 v16bf;
typedef __attribute__((ext_vector_type(8)))  float  v8f;

// ---------------- workspace layout (bf16 elements) ----------------
constexpr size_t SZ_WLA = (size_t)NT_S * KT_XA * 512;  // 13312
constexpr size_t SZ_G   = (size_t)NT_S * KT_S * 512;   // 46592
constexpr size_t SZ_H   = (size_t)2 * KT_S * 512;      // 7168
constexpr size_t SZ_WQ1 = (size_t)NT_S * KT_Q * 512;   // 259584

constexpr size_t OFF_WLA = 0;
constexpr size_t OFF_WIH = OFF_WLA + SZ_WLA;           // 3 gate chunks
constexpr size_t OFF_WHH = OFF_WIH + 3 * SZ_G;
constexpr size_t OFF_WP1 = OFF_WHH + 3 * SZ_G;
constexpr size_t OFF_WPM = OFF_WP1 + SZ_G;
constexpr size_t OFF_WPS = OFF_WPM + SZ_H;
constexpr size_t OFF_WQ1 = OFF_WPS + SZ_H;
constexpr size_t OFF_WQM = OFF_WQ1 + SZ_WQ1;
constexpr size_t OFF_WQS = OFF_WQM + SZ_H;

// ---------------- weight prep: f32 [N,K] row-major -> fragment-native bf16 ----
__global__ void prep_weight(const float* __restrict__ src, bf16_t* __restrict__ dst,
                            int N, int K, int NT, int KT) {
  int total = NT * KT * 512;
  for (int f = blockIdx.x * blockDim.x + threadIdx.x; f < total;
       f += gridDim.x * blockDim.x) {
    int tile = f >> 9;
    int rem  = f & 511;
    int lane = rem >> 4;
    int slot = rem & 15;
    int kt = tile % KT;
    int nt = tile / KT;
    int n = nt * 16 + (lane & 15);
    int k = kt * 32 + ((lane >> 4) << 3) + (slot & 7) + ((slot >> 3) << 4);
    float v = (n < N && k < K) ? src[(size_t)n * K + k] : 0.0f;
    dst[f] = (bf16_t)v;
  }
}

// ---------------- fragment loads ----------------
__device__ __forceinline__ v16bf load_a(const bf16_t* __restrict__ base, int ld,
                                        int kt, int lane) {
  const bf16_t* p = base + (lane & 15) * ld + kt * 32 + ((lane >> 4) << 3);
  v16bf r;
#pragma unroll
  for (int i = 0; i < 8; ++i) r[i] = p[i];
#pragma unroll
  for (int i = 0; i < 8; ++i) r[8 + i] = p[16 + i];
  return r;
}

__device__ __forceinline__ v16bf load_b(const bf16_t* __restrict__ w, int nt,
                                        int kt, int KT, int lane) {
  const bf16_t* p = w + (((size_t)(nt * KT + kt) << 5) + (size_t)lane) * 16;
  v16bf r;
#pragma unroll
  for (int i = 0; i < 16; ++i) r[i] = p[i];
  return r;
}

__device__ __forceinline__ v8f wmma_bf(v16bf a, v16bf b, v8f c) {
  return __builtin_amdgcn_wmma_f32_16x16x32_bf16(false, a, false, b, (short)0, c,
                                                 false, false);
}

// CHAINS==1: single pipelined D->C accumulation chain (no final add, no hazard)
// CHAINS==2: two chains for long-K GEMMs
template <int KT, int UNROLL, int CHAINS>
__device__ __forceinline__ v8f gemm_tile(const bf16_t* __restrict__ A, int ld,
                                         const bf16_t* __restrict__ W, int nt,
                                         int lane) {
  if constexpr (CHAINS == 1) {
    v8f acc = {};
#pragma unroll UNROLL
    for (int kt = 0; kt < KT; ++kt) {
      acc = wmma_bf(load_a(A, ld, kt, lane), load_b(W, nt, kt, KT, lane), acc);
    }
    return acc;
  } else {
    v8f acc0 = {}, acc1 = {};
#pragma unroll UNROLL
    for (int kt = 0; kt < KT; kt += 2) {
      acc0 = wmma_bf(load_a(A, ld, kt, lane), load_b(W, nt, kt, KT, lane), acc0);
      if (kt + 1 < KT)
        acc1 = wmma_bf(load_a(A, ld, kt + 1, lane),
                       load_b(W, nt, kt + 1, KT, lane), acc1);
    }
    return acc0 + acc1;
  }
}

__device__ __forceinline__ float sigmoidf_(float x) {
  return 1.0f / (1.0f + __expf(-x));
}
__device__ __forceinline__ float stdclampf_(float x) {
  float sp = fmaxf(x, 0.0f) + log1pf(__expf(-fabsf(x)));
  sp += 0.1f;
  return fminf(fmaxf(sp, 1e-6f), 10.0f);
}

// stage u_t -> s_sa[.,30..35] and e_{t+1} -> s_hq[.,200..1223]; prefetch e_{t+2}
__device__ __forceinline__ void stage_step(const float* __restrict__ e,
                                           const float* __restrict__ u,
                                           bf16_t* __restrict__ s_sa,
                                           bf16_t* __restrict__ s_hq, int m0,
                                           int t, int tid0, int nthreads) {
  for (int i = tid0; i < 16 * ACT; i += nthreads) {
    int m = i / ACT, k = i % ACT;
    s_sa[m * XAK + LATENT + k] =
        (bf16_t)u[((size_t)(m0 + m) * TT + t) * ACT + k];
  }
  for (int i = tid0; i < 16 * (EMBED / 4); i += nthreads) {
    int m = i >> 8;
    int j4 = (i & 255) << 2;
    const float4 v = *(const float4*)&e[((size_t)(m0 + m) * (TT + 1) + (t + 1)) *
                                            EMBED + j4];
    bf16_t* d = &s_hq[m * QKP + STATE + j4];
    d[0] = (bf16_t)v.x;
    d[1] = (bf16_t)v.y;
    d[2] = (bf16_t)v.z;
    d[3] = (bf16_t)v.w;
  }
  if (t + 2 <= TT) {
    for (int i = tid0; i < 1024; i += nthreads) {
      int m = i >> 6, line = i & 63;
      __builtin_prefetch(
          &e[((size_t)(m0 + m) * (TT + 1) + (t + 2)) * EMBED + line * 16], 0, 1);
    }
  }
}

// ---------------- main persistent-recurrence kernel ----------------
__global__ __launch_bounds__(256) void rssm_kernel(
    const float* __restrict__ e, const float* __restrict__ u,
    const float* __restrict__ b_la, const float* __restrict__ bih,
    const float* __restrict__ bhh, const float* __restrict__ b_p1,
    const float* __restrict__ b_pm, const float* __restrict__ b_ps,
    const float* __restrict__ b_q1, const float* __restrict__ b_qm,
    const float* __restrict__ b_qs, const bf16_t* __restrict__ wb,
    float* __restrict__ out) {
  __shared__ __align__(16) bf16_t s_h0[16 * SKP];  // h ping / zp scratch
  __shared__ __align__(16) bf16_t s_h1[16 * SKP];  // h pong / zp scratch
  __shared__ __align__(16) bf16_t s_xb[16 * SKP];  // x, later zq trunk
  __shared__ __align__(16) bf16_t s_sa[16 * XAK];  // concat(s,a)
  __shared__ __align__(16) bf16_t s_hq[16 * QKP];  // concat(h,e)

  const int tid = threadIdx.x;
  const int lane = tid & 31;
  const int w = tid >> 5;  // 8 waves
  const int m0 = blockIdx.x * 16;
  const int nloc = lane & 15;
  const int mrow = (lane >> 4) << 3;  // C rows = mrow + r

  float* out_states = out;
  float* out_pm = out + (size_t)TT * BB * STATE;
  float* out_ps = out_pm + (size_t)TT * BB * LATENT;
  float* out_qm = out_ps + (size_t)TT * BB * LATENT;
  float* out_qs = out_qm + (size_t)TT * BB * LATENT;

  const bf16_t* Wla = wb + OFF_WLA;
  const bf16_t* WihR = wb + OFF_WIH;
  const bf16_t* WihZ = WihR + SZ_G;
  const bf16_t* WihN = WihZ + SZ_G;
  const bf16_t* WhhR = wb + OFF_WHH;
  const bf16_t* WhhZ = WhhR + SZ_G;
  const bf16_t* WhhN = WhhZ + SZ_G;
  const bf16_t* Wp1 = wb + OFF_WP1;
  const bf16_t* Wq1 = wb + OFF_WQ1;

  // zero LDS
  for (int i = tid; i < 16 * SKP; i += 256) {
    s_h0[i] = (bf16_t)0.0f;
    s_h1[i] = (bf16_t)0.0f;
    s_xb[i] = (bf16_t)0.0f;
  }
  for (int i = tid; i < 16 * XAK; i += 256) s_sa[i] = (bf16_t)0.0f;
  for (int i = tid; i < 16 * QKP; i += 256) s_hq[i] = (bf16_t)0.0f;
  __syncthreads();

  // pre-stage step 0 inputs with all threads
  stage_step(e, u, s_sa, s_hq, m0, 0, tid, 256);

  // static tile ownership: nt0 = w, nt1 = w+8 (nt1 valid only for w<5)
  const int nt0 = w;
  const int nt1 = w + 8;
  const bool t1ok = (nt1 < NT_S);
  const int n0 = nt0 * 16 + nloc;  // always < STATE
  const int n1 = nt1 * 16 + nloc;
  const bool v1 = t1ok && (n1 < STATE);

  // ---- hoist all per-lane biases out of the recurrence ----
  const float bla0 = b_la[n0];
  const float bla1 = v1 ? b_la[n1] : 0.0f;
  const float bir0 = bih[n0], biz0 = bih[STATE + n0], bin0 = bih[2 * STATE + n0];
  const float bhr0 = bhh[n0], bhz0 = bhh[STATE + n0], bhn0 = bhh[2 * STATE + n0];
  const float bir1 = v1 ? bih[n1] : 0.0f;
  const float biz1 = v1 ? bih[STATE + n1] : 0.0f;
  const float bin1 = v1 ? bih[2 * STATE + n1] : 0.0f;
  const float bhr1 = v1 ? bhh[n1] : 0.0f;
  const float bhz1 = v1 ? bhh[STATE + n1] : 0.0f;
  const float bhn1 = v1 ? bhh[2 * STATE + n1] : 0.0f;
  const float bp10 = b_p1[n0];
  const float bp11 = v1 ? b_p1[n1] : 0.0f;
  const float bq10 = b_q1[n0];
  const float bq11 = v1 ? b_q1[n1] : 0.0f;

  // head job: head = w>>1 (0=pm,1=ps,2=qm,3=qs), tile = w&1
  const int head = w >> 1;
  const int hnt = w & 1;
  const int nh = hnt * 16 + nloc;
  const bool hvalid = (nh < LATENT);
  const bool h_is_std = (head & 1);
  const bool h_is_qm = (head == 2);
  const bf16_t* Whead =
      wb + (head == 0 ? OFF_WPM : head == 1 ? OFF_WPS : head == 2 ? OFF_WQM
                                                                  : OFF_WQS);
  float* out_head = (head == 0) ? out_pm : (head == 1) ? out_ps
                    : (head == 2) ? out_qm : out_qs;
  const float bhd =
      hvalid ? (head == 0 ? b_pm[nh] : head == 1 ? b_ps[nh] : head == 2 ? b_qm[nh]
                                                                        : b_qs[nh])
             : 0.0f;

  v8f hreg0 = {};  // h_old for owned tile 0
  v8f hreg1 = {};  // h_old for owned tile 1
  __syncthreads();

  for (int t = 0; t < TT; ++t) {
    // step-parity ping-pong for h
    bf16_t* hb_cur = (t & 1) ? s_h1 : s_h0;  // holds h_{t-1}; becomes zp buffer
    bf16_t* hb_nxt = (t & 1) ? s_h0 : s_h1;  // receives h_t

    // ---- phase 1: x = relu([s|a] @ Wla^T + b_la) -> s_xb
    {
      v8f acc = gemm_tile<KT_XA, 2, 1>(s_sa, XAK, Wla, nt0, lane);
#pragma unroll
      for (int r = 0; r < 8; ++r)
        s_xb[(mrow + r) * SKP + n0] = (bf16_t)fmaxf(acc[r] + bla0, 0.0f);
      if (t1ok) {
        v8f a2 = gemm_tile<KT_XA, 2, 1>(s_sa, XAK, Wla, nt1, lane);
#pragma unroll
        for (int r = 0; r < 8; ++r)
          s_xb[(mrow + r) * SKP + n1] = (bf16_t)fmaxf(a2[r] + bla1, 0.0f);
      }
    }
    __syncthreads();

    // ---- phase 2: GRU per owned tile; reads hb_cur/s_xb, writes hb_nxt.
    //      No mid-phase barrier needed thanks to the ping-pong buffer.
#pragma unroll
    for (int tile = 0; tile < 2; ++tile) {
      if (tile && !t1ok) break;
      const int nt = tile ? nt1 : nt0;
      v8f ar = {}, az = {}, an = {}, ahn = {};
#pragma unroll
      for (int kt = 0; kt < KT_S; ++kt) {
        v16bf ah = load_a(hb_cur, SKP, kt, lane);
        v16bf ax = load_a(s_xb, SKP, kt, lane);
        ar = wmma_bf(ah, load_b(WhhR, nt, kt, KT_S, lane), ar);
        ar = wmma_bf(ax, load_b(WihR, nt, kt, KT_S, lane), ar);
        az = wmma_bf(ah, load_b(WhhZ, nt, kt, KT_S, lane), az);
        az = wmma_bf(ax, load_b(WihZ, nt, kt, KT_S, lane), az);
        ahn = wmma_bf(ah, load_b(WhhN, nt, kt, KT_S, lane), ahn);
        an = wmma_bf(ax, load_b(WihN, nt, kt, KT_S, lane), an);
      }
      const int n = tile ? n1 : n0;
      const bool valid = tile ? v1 : true;
      const float bir = tile ? bir1 : bir0, biz = tile ? biz1 : biz0;
      const float bin = tile ? bin1 : bin0, bhr = tile ? bhr1 : bhr0;
      const float bhz = tile ? bhz1 : bhz0, bhn = tile ? bhn1 : bhn0;
#pragma unroll
      for (int r = 0; r < 8; ++r) {
        float rg = sigmoidf_(ar[r] + bir + bhr);
        float zg = sigmoidf_(az[r] + biz + bhz);
        float ng = tanhf(an[r] + bin + rg * (ahn[r] + bhn));
        float hold = tile ? hreg1[r] : hreg0[r];
        float hnew = (1.0f - zg) * ng + zg * hold;
        if (tile) hreg1[r] = hnew; else hreg0[r] = hnew;
        int m = mrow + r;
        bf16_t hb = (bf16_t)hnew;
        hb_nxt[m * SKP + n] = hb;
        if (valid) {
          s_hq[m * QKP + n] = hb;
          out_states[((size_t)t * BB + (m0 + m)) * STATE + n] = hnew;
        }
      }
    }
    __syncthreads();

    // ---- phase 3: BOTH trunks. zq -> s_xb (x dead), zp -> hb_cur (old h dead)
    {
      v8f aq = gemm_tile<KT_Q, 3, 2>(s_hq, QKP, Wq1, nt0, lane);
#pragma unroll
      for (int r = 0; r < 8; ++r)
        s_xb[(mrow + r) * SKP + n0] = (bf16_t)fmaxf(aq[r] + bq10, 0.0f);
      v8f ap = gemm_tile<KT_S, 7, 1>(hb_nxt, SKP, Wp1, nt0, lane);
#pragma unroll
      for (int r = 0; r < 8; ++r)
        hb_cur[(mrow + r) * SKP + n0] = (bf16_t)fmaxf(ap[r] + bp10, 0.0f);
      if (t1ok) {
        v8f aq1 = gemm_tile<KT_Q, 3, 2>(s_hq, QKP, Wq1, nt1, lane);
#pragma unroll
        for (int r = 0; r < 8; ++r)
          s_xb[(mrow + r) * SKP + n1] = (bf16_t)fmaxf(aq1[r] + bq11, 0.0f);
        v8f ap1 = gemm_tile<KT_S, 7, 1>(hb_nxt, SKP, Wp1, nt1, lane);
#pragma unroll
        for (int r = 0; r < 8; ++r)
          hb_cur[(mrow + r) * SKP + n1] = (bf16_t)fmaxf(ap1[r] + bp11, 0.0f);
      }
    }
    __syncthreads();

    // ---- phase 4: 8 head GEMMs (one per wave) + next-step staging
    {
      const bf16_t* Atrunk = (head < 2) ? hb_cur : s_xb;  // zp : zq
      v8f acc = gemm_tile<KT_S, 7, 1>(Atrunk, SKP, Whead, hnt, lane);
      if (hvalid) {
#pragma unroll
        for (int r = 0; r < 8; ++r) {
          float v = acc[r] + bhd;
          if (h_is_std) v = stdclampf_(v);
          int m = mrow + r;
          out_head[((size_t)t * BB + (m0 + m)) * LATENT + nh] = v;
          if (h_is_qm) s_sa[m * XAK + nh] = (bf16_t)v;  // s_{t+1} = qm
        }
      }
      if (t + 1 < TT) stage_step(e, u, s_sa, s_hq, m0, t + 1, tid, 256);
    }
    __syncthreads();
  }
}

// ---------------- host launch ----------------
extern "C" void kernel_launch(void* const* d_in, const int* in_sizes, int n_in,
                              void* d_out, int out_size, void* d_ws,
                              size_t ws_size, hipStream_t stream) {
  const float* e   = (const float*)d_in[0];
  const float* u   = (const float*)d_in[1];
  const float* Wla = (const float*)d_in[2];
  const float* bla = (const float*)d_in[3];
  const float* Wih = (const float*)d_in[4];
  const float* Whh = (const float*)d_in[5];
  const float* bih = (const float*)d_in[6];
  const float* bhh = (const float*)d_in[7];
  const float* Wp1 = (const float*)d_in[8];
  const float* bp1 = (const float*)d_in[9];
  const float* Wpm = (const float*)d_in[10];
  const float* bpm = (const float*)d_in[11];
  const float* Wps = (const float*)d_in[12];
  const float* bps = (const float*)d_in[13];
  const float* Wq1 = (const float*)d_in[14];
  const float* bq1 = (const float*)d_in[15];
  const float* Wqm = (const float*)d_in[16];
  const float* bqm = (const float*)d_in[17];
  const float* Wqs = (const float*)d_in[18];
  const float* bqs = (const float*)d_in[19];
  bf16_t* wb = (bf16_t*)d_ws;

  auto prep = [&](const float* src, size_t off, int N, int K, int NT, int KT) {
    int total = NT * KT * 512;
    int blocks = (total + 255) / 256;
    prep_weight<<<blocks, 256, 0, stream>>>(src, wb + off, N, K, NT, KT);
  };

  prep(Wla, OFF_WLA, STATE, LATENT + ACT, NT_S, KT_XA);
  for (int g = 0; g < 3; ++g) {
    prep(Wih + (size_t)g * STATE * STATE, OFF_WIH + g * SZ_G, STATE, STATE,
         NT_S, KT_S);
    prep(Whh + (size_t)g * STATE * STATE, OFF_WHH + g * SZ_G, STATE, STATE,
         NT_S, KT_S);
  }
  prep(Wp1, OFF_WP1, STATE, STATE, NT_S, KT_S);
  prep(Wpm, OFF_WPM, LATENT, STATE, 2, KT_S);
  prep(Wps, OFF_WPS, LATENT, STATE, 2, KT_S);
  prep(Wq1, OFF_WQ1, STATE, STATE + EMBED, NT_S, KT_Q);
  prep(Wqm, OFF_WQM, LATENT, STATE, 2, KT_S);
  prep(Wqs, OFF_WQS, LATENT, STATE, 2, KT_S);

  rssm_kernel<<<BB / 16, 256, 0, stream>>>(e, u, bla, bih, bhh, bp1, bpm, bps,
                                           bq1, bqm, bqs, wb, (float*)d_out);
}
